// ChangeDecoder_73186242724383
// MI455X (gfx1250) — matrix operand order repaired and checked by reading, not compiled, not run on hardware
//
#include <hip/hip_runtime.h>
#include <hip/hip_bf16.h>
#include <math.h>

// MI455X (gfx1250) implementation.
// All matmuls (1x1 convs, in_proj, x_proj, dt_proj, out_proj, MLP, fuse,
// 3x3 convs via im2col) run through one bf16 WMMA GEMM kernel using
// V_WMMA_F32_16X16X32_BF16 with f32 accumulate, 32x32 macro-tile per wave
// (4 independent accumulators -> back-to-back XDL issue), fused epilogues.
// All problem shapes are padded so K%32==0 and lda/ldb%8==0: interior tiles
// run a branch-free load+wmma loop; only M/N edge tiles take row-guarded
// vector loads. The SS2D selective scan is sequential f32 VALU w/ prefetch.

typedef __bf16 bf16_t;
typedef __attribute__((ext_vector_type(16))) __bf16 v16bf;
typedef __attribute__((ext_vector_type(8)))  __bf16 v8bf;
typedef __attribute__((ext_vector_type(8)))  float  v8f;

#define LN_EPS 1e-5f

__device__ __forceinline__ float sigmoidf_(float x){ return 1.f/(1.f+__expf(-x)); }

__device__ __forceinline__ v16bf zero16(){
  v16bf z;
  #pragma unroll
  for (int e = 0; e < 16; e++) z[e] = (bf16_t)0.f;
  return z;
}
__device__ __forceinline__ v16bf load16(const bf16_t* __restrict__ p, int off0, int off1){
  v8bf lo = *(const v8bf*)(p + off0);
  v8bf hi = *(const v8bf*)(p + off1);
  return __builtin_shufflevector(lo, hi, 0,1,2,3,4,5,6,7,8,9,10,11,12,13,14,15);
}
__device__ __forceinline__ v16bf load16g(const bf16_t* __restrict__ p, bool ok, int off0, int off1){
  v16bf r = zero16();
  if (ok) r = load16(p, off0, off1);
  return r;
}

// ------------------------------------------------------------------
// conversion / padding kernels
// ------------------------------------------------------------------
__global__ void k_f32_to_bf16(const float* __restrict__ s, bf16_t* __restrict__ d, long n){
  long i = (long)blockIdx.x*blockDim.x + threadIdx.x;
  if (i < n) d[i] = (bf16_t)s[i];
}
// s[K,N] row-major -> d[N,K] row-major (weight transpose for WMMA B-matrix)
__global__ void k_f32_to_bf16_T(const float* __restrict__ s, bf16_t* __restrict__ d, int K, int N){
  long i = (long)blockIdx.x*blockDim.x + threadIdx.x;
  long tot = (long)K*N;
  if (i < tot){ int k = (int)(i / N), n = (int)(i % N); d[(long)n*K + k] = (bf16_t)s[i]; }
}
// copy first ncols cols of f32 [rows, ld_src] -> bf16 [rows, ld_dst], zero-pad rest
__global__ void k_pad_cols_bf16(const float* __restrict__ s, int ld_src, int ncols,
                                bf16_t* __restrict__ d, int ld_dst, long rows){
  long i = (long)blockIdx.x*blockDim.x + threadIdx.x;
  long tot = rows * ld_dst;
  if (i >= tot) return;
  long r = i / ld_dst; int c = (int)(i % ld_dst);
  d[i] = (c < ncols) ? (bf16_t)s[r*(long)ld_src + c] : (bf16_t)0.f;
}
// x_proj_w (4,24,256) f32 -> (4,32,256) bf16, zero rows 24..31
__global__ void k_prep_wxp(const float* __restrict__ s, bf16_t* __restrict__ d){
  long i = (long)blockIdx.x*blockDim.x + threadIdx.x;
  const long tot = 4L*32*256;
  if (i >= tot) return;
  int j = (int)(i % 256);
  int r = (int)((i / 256) % 32);
  int k = (int)(i / (32*256));
  d[i] = (r < 24) ? (bf16_t)s[((long)k*24 + r)*256 + j] : (bf16_t)0.f;
}

// ------------------------------------------------------------------
// WMMA GEMM:  C[M,N] = act( (A[M,K] @ B[N,K]^T + bias)*scale + shift + res )
// Requirements (guaranteed by host): K % 32 == 0, lda % 8 == 0, ldb % 8 == 0.
// One wave computes a 32x32 macro-tile = 2x2 WMMA tiles, 4 independent accs.
// Store: transC=0 -> out[m*ldc+n] ; transC=1 -> out[n*ldc+m] (NCHW store).
// act: 0 none, 1 relu, 2 gelu(exact), 3 silu
// ------------------------------------------------------------------
__device__ __forceinline__ void gemm_epilogue(
    v8f acc, int m0, int n0, int lane, int M, int N,
    float* __restrict__ outf, bf16_t* __restrict__ outb, int ldc, int transC,
    const float* __restrict__ bias, const float* __restrict__ scale, float smul,
    const float* __restrict__ shift, const float* __restrict__ res, int act)
{
  const int rit = lane & 15;
  const int n = n0 + rit;
  const int mb = m0 + ((lane >> 4) << 3);
  if (n >= N) return;
  const float bs = bias  ? bias[n]         : 0.f;
  const float sc = scale ? scale[n] * smul : 1.f;
  const float sh = shift ? shift[n]        : 0.f;
  #pragma unroll
  for (int v = 0; v < 8; v++){
    const int m = mb + v;
    if (m >= M) break;
    float tv = acc[v] + bs;
    tv = tv * sc + sh;
    const long idx = transC ? ((long)n * ldc + m) : ((long)m * ldc + n);
    if (res) tv += res[idx];
    if      (act == 1) tv = fmaxf(tv, 0.f);
    else if (act == 2) tv = 0.5f * tv * (1.f + erff(tv * 0.70710678118654752f));
    else if (act == 3) tv = tv * sigmoidf_(tv);
    if (outf) outf[idx] = tv;
    if (outb) outb[idx] = (bf16_t)tv;
  }
}

__global__ void __launch_bounds__(256)
k_gemm_wmma(const bf16_t* __restrict__ A, int lda,
            const bf16_t* __restrict__ B, int ldb,
            int M, int N, int K,
            float* __restrict__ outf, bf16_t* __restrict__ outb,
            int ldc, int transC,
            const float* __restrict__ bias,
            const float* __restrict__ scale, float scale_mul,
            const float* __restrict__ shift,
            const float* __restrict__ res, int act)
{
  const int lane   = threadIdx.x & 31;
  const int wave   = blockIdx.x * (blockDim.x >> 5) + (threadIdx.x >> 5);
  const int nwaves = gridDim.x * (blockDim.x >> 5);
  const int tm = (M + 31) >> 5, tn = (N + 31) >> 5;   // 32x32 macro tiles
  const int rit  = lane & 15;
  const int kb_a = (lane >> 4) << 3;     // A K-offset group: 0 or 8
  const int kg_b = (lane >> 4) << 4;     // unused; B loads use kb_a too (0 or 8)

  for (int t = wave; t < tm * tn; t += nwaves){
    const int m0 = (t / tn) << 5;
    const int n0 = (t % tn) << 5;
    v8f acc00 = {}, acc01 = {}, acc10 = {}, acc11 = {};

    const bf16_t* Ar0 = A + (long)(m0 + rit) * lda;
    const bf16_t* Ar1 = Ar0 + (long)16 * lda;
    const bf16_t* Br0 = B + (long)(n0 + rit) * ldb;
    const bf16_t* Br1 = Br0 + (long)16 * ldb;

    if ((m0 + 32 <= M) && (n0 + 32 <= N)){
      // interior macro-tile: branch-free vector loads + 4 independent WMMAs
      for (int k0 = 0; k0 < K; k0 += 32){
        const int a_o0 = k0 + kb_a, a_o1 = k0 + 16 + kb_a;
        const int b_o0 = k0 + kb_a, b_o1 = k0 + 16 + kb_a;
        v16bf a0 = load16(Ar0, a_o0, a_o1);
        v16bf a1 = load16(Ar1, a_o0, a_o1);
        v16bf b0 = load16(Br0, b_o0, b_o1);
        v16bf b1 = load16(Br1, b_o0, b_o1);
        acc00 = __builtin_amdgcn_wmma_f32_16x16x32_bf16(false, a0, false, b0, (short)0, acc00, false, false);
        acc01 = __builtin_amdgcn_wmma_f32_16x16x32_bf16(false, a0, false, b1, (short)0, acc01, false, false);
        acc10 = __builtin_amdgcn_wmma_f32_16x16x32_bf16(false, a1, false, b0, (short)0, acc10, false, false);
        acc11 = __builtin_amdgcn_wmma_f32_16x16x32_bf16(false, a1, false, b1, (short)0, acc11, false, false);
      }
    } else {
      // edge macro-tile: row-guarded vector loads (K always in range)
      const bool aok0 = (m0 + rit)      < M;
      const bool aok1 = (m0 + 16 + rit) < M;
      const bool bok0 = (n0 + rit)      < N;
      const bool bok1 = (n0 + 16 + rit) < N;
      for (int k0 = 0; k0 < K; k0 += 32){
        const int a_o0 = k0 + kb_a, a_o1 = k0 + 16 + kb_a;
        const int b_o0 = k0 + kb_a, b_o1 = k0 + 16 + kb_a;
        v16bf a0 = load16g(Ar0, aok0, a_o0, a_o1);
        v16bf a1 = load16g(Ar1, aok1, a_o0, a_o1);
        v16bf b0 = load16g(Br0, bok0, b_o0, b_o1);
        v16bf b1 = load16g(Br1, bok1, b_o0, b_o1);
        acc00 = __builtin_amdgcn_wmma_f32_16x16x32_bf16(false, a0, false, b0, (short)0, acc00, false, false);
        acc01 = __builtin_amdgcn_wmma_f32_16x16x32_bf16(false, a0, false, b1, (short)0, acc01, false, false);
        acc10 = __builtin_amdgcn_wmma_f32_16x16x32_bf16(false, a1, false, b0, (short)0, acc10, false, false);
        acc11 = __builtin_amdgcn_wmma_f32_16x16x32_bf16(false, a1, false, b1, (short)0, acc11, false, false);
      }
    }

    gemm_epilogue(acc00, m0,      n0,      lane, M, N, outf, outb, ldc, transC, bias, scale, scale_mul, shift, res, act);
    gemm_epilogue(acc01, m0,      n0 + 16, lane, M, N, outf, outb, ldc, transC, bias, scale, scale_mul, shift, res, act);
    gemm_epilogue(acc10, m0 + 16, n0,      lane, M, N, outf, outb, ldc, transC, bias, scale, scale_mul, shift, res, act);
    gemm_epilogue(acc11, m0 + 16, n0 + 16, lane, M, N, outf, outb, ldc, transC, bias, scale, scale_mul, shift, res, act);
  }
}

// ------------------------------------------------------------------
// LayerNorm over rows of D (wave per row) with optional SiLU gate.
// ------------------------------------------------------------------
__global__ void k_ln(const float* __restrict__ x, int L, int D,
                     const float* __restrict__ g, const float* __restrict__ b,
                     float* __restrict__ outf, bf16_t* __restrict__ outb,
                     const float* __restrict__ gate, int gate_ld, int gate_off)
{
  const int row = blockIdx.x * (blockDim.x >> 5) + (threadIdx.x >> 5);
  if (row >= L) return;
  const int lane = threadIdx.x & 31;
  const float* xr = x + (long)row * D;
  float s = 0.f, s2 = 0.f;
  for (int i = lane; i < D; i += 32){ float v = xr[i]; s += v; s2 += v * v; }
  for (int off = 16; off; off >>= 1){ s += __shfl_xor(s, off, 32); s2 += __shfl_xor(s2, off, 32); }
  const float mean = s / D;
  float var = s2 / D - mean * mean; var = fmaxf(var, 0.f);
  const float inv = rsqrtf(var + LN_EPS);
  for (int i = lane; i < D; i += 32){
    float v = (xr[i] - mean) * inv * g[i] + b[i];
    if (gate){
      const float z = gate[(long)row * gate_ld + gate_off + i];
      v *= z * sigmoidf_(z);
    }
    if (outf) outf[(long)row * D + i] = v;
    if (outb) outb[(long)row * D + i] = (bf16_t)v;
  }
}

// ------------------------------------------------------------------
// depthwise 3x3 conv (SAME) + bias + SiLU on xi = xz[:, :256]; NHWC out
// ------------------------------------------------------------------
__global__ void k_dwconv_silu(const float* __restrict__ xz, const float* __restrict__ w,
                              const float* __restrict__ b, float* __restrict__ xc,
                              int H, int W)
{
  long i = (long)blockIdx.x*blockDim.x + threadIdx.x;
  const long tot = (long)H * W * 256;
  if (i >= tot) return;
  const int c = (int)(i & 255);
  const long l = i >> 8;
  const int h = (int)(l / W), wx = (int)(l % W);
  float acc = b[c];
  for (int dy = 0; dy < 3; dy++)
    for (int dx = 0; dx < 3; dx++){
      const int yy = h + dy - 1, xx = wx + dx - 1;
      if (yy >= 0 && yy < H && xx >= 0 && xx < W)
        acc += xz[((long)yy * W + xx) * 512 + c] * w[c * 9 + dy * 3 + dx];
    }
  xc[(long)l * 256 + c] = acc * sigmoidf_(acc);
}

// build xs bf16 [4][L][256]: dir0 row-major, dir1 col-major, dir2/3 flipped
__global__ void k_prep_xs(const float* __restrict__ xc, bf16_t* __restrict__ xs, int H, int W)
{
  const long L = (long)H * W;
  const long tot = 4 * L * 256;
  for (long i = (long)blockIdx.x*blockDim.x + threadIdx.x; i < tot; i += (long)gridDim.x*blockDim.x){
    const int d = (int)(i & 255);
    const long t = i >> 8;
    const long pos = t % L;
    const int k = (int)(t / L);
    long j = (k & 2) ? (L - 1 - pos) : pos;
    long l;
    if (k & 1){ const int h = (int)(j % H), w = (int)(j / H); l = (long)h * W + w; }
    else l = j;
    xs[i] = (bf16_t)xc[l * 256 + d];
  }
}

// ------------------------------------------------------------------
// SS2D sequential scan: thread per (direction k, channel d); 8 states in regs
// xdbl rows are stride 32: [0..7]=dt-rank (unused here), [8..15]=B, [16..23]=C
// ------------------------------------------------------------------
__global__ void k_scan(const float* __restrict__ dts, const bf16_t* __restrict__ xs,
                       const float* __restrict__ xdbl, const float* __restrict__ Alog,
                       const float* __restrict__ Ds, const float* __restrict__ dtb,
                       float* __restrict__ Y, int L)
{
  const int tid = blockIdx.x * blockDim.x + threadIdx.x;
  if (tid >= 4 * 256) return;
  const int k = tid >> 8, d = tid & 255;
  float Ac[8];
  for (int n = 0; n < 8; n++) Ac[n] = -__expf(Alog[((long)(k * 256 + d)) * 8 + n]);
  const float Dd = Ds[k * 256 + d];
  const float bias = dtb[k * 256 + d];
  float h[8];
  #pragma unroll
  for (int n = 0; n < 8; n++) h[n] = 0.f;
  const float*  dts_k = dts  + (long)k * L * 256;
  const bf16_t* xs_k  = xs   + (long)k * L * 256;
  const float*  xd_k  = xdbl + (long)k * L * 32;
  float* Yk = Y + ((long)k * 256 + d) * L;
  for (int i = 0; i < L; i++){
    const float* row = xd_k + (long)i * 32;
    __builtin_prefetch(row + 32 * 8, 0, 1);                    // global_prefetch ahead
    __builtin_prefetch(dts_k + (long)(i + 8) * 256 + d, 0, 1);
    float dtv = dts_k[(long)i * 256 + d] + bias;
    dtv = (dtv > 20.f) ? dtv : log1pf(__expf(dtv));            // softplus
    const float u  = (float)xs_k[(long)i * 256 + d];
    const float du = dtv * u;
    float y = u * Dd;
    #pragma unroll
    for (int n = 0; n < 8; n++){
      const float a = __expf(dtv * Ac[n]);
      h[n] = a * h[n] + du * row[8 + n];
      y += h[n] * row[16 + n];
    }
    Yk[i] = y;
  }
}

// combine 4 scan directions -> ym [L,256] row-major pixel order
__global__ void k_combine(const float* __restrict__ Y, float* __restrict__ ym, int H, int W)
{
  const long L = (long)H * W;
  const long tot = L * 256;
  long i = (long)blockIdx.x*blockDim.x + threadIdx.x;
  if (i >= tot) return;
  const int d = (int)(i & 255);
  const long l = i >> 8;
  const int h = (int)(l / W), w = (int)(l % W);
  const long lc = (long)w * H + h;
  ym[i] = Y[((long)(0 * 256 + d)) * L + l]
        + Y[((long)(2 * 256 + d)) * L + (L - 1 - l)]
        + Y[((long)(1 * 256 + d)) * L + lc]
        + Y[((long)(3 * 256 + d)) * L + (L - 1 - lc)];
}

// ------------------------------------------------------------------
// stage input prep (concat-channel / interleave-W / concat-W), NCHW f32 -> bf16 [L,cin']
// ------------------------------------------------------------------
__global__ void k_prep_A(const float* __restrict__ pre, const float* __restrict__ post,
                         bf16_t* __restrict__ A, int H, int W, int cin, int mode)
{
  long i = (long)blockIdx.x*blockDim.x + threadIdx.x;
  const long HW = (long)H * W;
  if (mode == 0){
    const long tot = HW * 2 * cin;
    if (i >= tot) return;
    const int c2 = (int)(i % (2 * cin));
    const long l = i / (2 * cin);
    const float v = (c2 < cin) ? pre[(long)c2 * HW + l] : post[(long)(c2 - cin) * HW + l];
    A[i] = (bf16_t)v;
  } else {
    const long tot = (long)H * 2 * W * cin;
    if (i >= tot) return;
    const int c = (int)(i % cin);
    const long l2 = i / cin;
    const int w2 = (int)(l2 % (2 * W));
    const int h  = (int)(l2 / (2 * W));
    const float* src; long l;
    if (mode == 1){ src = (w2 & 1) ? post : pre; l = (long)h * W + (w2 >> 1); }
    else { if (w2 < W){ src = pre; l = (long)h * W + w2; } else { src = post; l = (long)h * W + (w2 - W); } }
    A[i] = (bf16_t)src[(long)c * HW + l];
  }
}

// gather fuse input: 640 channels per pixel from q1/q2/q3 -> bf16 [HW,640]
__global__ void k_prep_fuse(const float* __restrict__ q1, const float* __restrict__ q2,
                            const float* __restrict__ q3, bf16_t* __restrict__ A, int H, int W)
{
  long i = (long)blockIdx.x*blockDim.x + threadIdx.x;
  const long tot = (long)H * W * 640;
  if (i >= tot) return;
  const int c640 = (int)(i % 640);
  const long l = i / 640;
  const int h = (int)(l / W), w = (int)(l % W);
  const int grp = c640 >> 7, c = c640 & 127;
  float v;
  const long W2 = 2L * W;
  switch (grp){
    case 0:  v = q1[l * 128 + c]; break;
    case 1:  v = q2[((long)h * W2 + 2 * w)     * 128 + c]; break;
    case 2:  v = q2[((long)h * W2 + 2 * w + 1) * 128 + c]; break;
    case 3:  v = q3[((long)h * W2 + w)         * 128 + c]; break;
    default: v = q3[((long)h * W2 + W + w)     * 128 + c]; break;
  }
  A[i] = (bf16_t)v;
}

// im2col: NCHW [128,H,W] -> bf16 [HW, 1152] (col = c*9 + dy*3 + dx), zero pad
__global__ void k_im2col(const float* __restrict__ x, bf16_t* __restrict__ A, int H, int W)
{
  long i = (long)blockIdx.x*blockDim.x + threadIdx.x;
  const long tot = (long)H * W * 1152;
  if (i >= tot) return;
  const int col = (int)(i % 1152);
  const long l = i / 1152;
  const int c = col / 9, r = col % 9, dy = r / 3, dx = r % 3;
  const int h = (int)(l / W) + dy - 1, w = (int)(l % W) + dx - 1;
  A[i] = (h >= 0 && h < H && w >= 0 && w < W) ? (bf16_t)x[((long)c * H + h) * W + w] : (bf16_t)0.f;
}

// bilinear (half-pixel) upsample of NCHW small[128,h,w] to [128,H,W] + big
__global__ void k_upsample_add(const float* __restrict__ small, int h, int w,
                               const float* __restrict__ big, float* __restrict__ out,
                               int H, int W)
{
  long i = (long)blockIdx.x*blockDim.x + threadIdx.x;
  const long tot = 128L * H * W;
  if (i >= tot) return;
  const int x = (int)(i % W);
  const long t = i / W;
  const int y = (int)(t % H);
  const int c = (int)(t / H);
  float fy = (y + 0.5f) * (float)h / (float)H - 0.5f;
  float fx = (x + 0.5f) * (float)w / (float)W - 0.5f;
  fy = fminf(fmaxf(fy, 0.f), (float)(h - 1));
  fx = fminf(fmaxf(fx, 0.f), (float)(w - 1));
  const int y0 = (int)fy, x0 = (int)fx;
  const int y1 = min(y0 + 1, h - 1), x1 = min(x0 + 1, w - 1);
  const float wy = fy - y0, wx = fx - x0;
  const float* s = small + (long)c * h * w;
  const float v = (1.f - wy) * ((1.f - wx) * s[(long)y0 * w + x0] + wx * s[(long)y0 * w + x1])
                +        wy  * ((1.f - wx) * s[(long)y1 * w + x0] + wx * s[(long)y1 * w + x1]);
  out[i] = v + big[i];
}

// ==================================================================
// host orchestration
// ==================================================================
static inline int cdiv(long a, int b){ return (int)((a + b - 1) / b); }

struct Ctx {
  hipStream_t st;
  void* const* din;
  // f32 scratch
  float *XRES, *XZ, *XC, *XDBL, *DTS, *Y, *YM, *X2, *Q1, *Q2, *Q3;
  float *P4, *P3, *P2, *PF, *T1, *T2;
  // bf16 scratch
  bf16_t *ABUF, *XS, *YG, *HM, *DTA;
  bf16_t *WCONV, *WINP, *WXP, *WDTP, *WOUTP, *WM1, *WM2, *WF, *WSM;
};

static void gemm(Ctx& c, const bf16_t* A, int lda, const bf16_t* B, int ldb,
                 int M, int N, int K, float* outf, bf16_t* outb, int ldc, int transC,
                 const float* bias, const float* scale, float smul, const float* shift,
                 const float* res, int act)
{
  const int tiles = ((M + 31) / 32) * ((N + 31) / 32);
  int blocks = (tiles + 7) / 8;
  if (blocks < 1) blocks = 1;
  if (blocks > 8192) blocks = 8192;
  k_gemm_wmma<<<blocks, 256, 0, c.st>>>(A, lda, B, ldb, M, N, K, outf, outb, ldc, transC,
                                        bias, scale, smul, shift, res, act);
}

static void cvt(Ctx& c, const float* s, bf16_t* d, long n){
  k_f32_to_bf16<<<cdiv(n, 256), 256, 0, c.st>>>(s, d, n);
}
static void cvtT(Ctx& c, const float* s, bf16_t* d, int K, int N){
  k_f32_to_bf16_T<<<cdiv((long)K * N, 256), 256, 0, c.st>>>(s, d, K, N);
}

// one st_block (1x1 conv + VSS block); A input already in c.ABUF as bf16 [L,cin]
static void run_st_block(Ctx& c, int pb, int H, int Wd, int cin, float* outQ)
{
  const float* conv_w     = (const float*)c.din[pb + 0];
  const float* conv_b     = (const float*)c.din[pb + 1];
  const float* norm_g     = (const float*)c.din[pb + 2];
  const float* norm_b     = (const float*)c.din[pb + 3];
  const float* in_proj_w  = (const float*)c.din[pb + 4];
  const float* dw_w       = (const float*)c.din[pb + 5];
  const float* dw_b       = (const float*)c.din[pb + 6];
  const float* x_proj_w   = (const float*)c.din[pb + 7];
  const float* dt_proj_w  = (const float*)c.din[pb + 8];
  const float* dt_bias    = (const float*)c.din[pb + 9];
  const float* A_log      = (const float*)c.din[pb + 10];
  const float* Ds         = (const float*)c.din[pb + 11];
  const float* out_norm_g = (const float*)c.din[pb + 12];
  const float* out_norm_b = (const float*)c.din[pb + 13];
  const float* out_proj_w = (const float*)c.din[pb + 14];
  const float* norm2_g    = (const float*)c.din[pb + 15];
  const float* norm2_b    = (const float*)c.din[pb + 16];
  const float* mlp_w1     = (const float*)c.din[pb + 17];
  const float* mlp_b1     = (const float*)c.din[pb + 18];
  const float* mlp_w2     = (const float*)c.din[pb + 19];
  const float* mlp_b2     = (const float*)c.din[pb + 20];

  const int L = H * Wd;

  // 1x1 conv: xres = A @ conv_w^T + b   (conv_w is already [N=128,K=cin])
  cvt(c, conv_w, c.WCONV, 128L * cin);
  gemm(c, c.ABUF, cin, c.WCONV, cin, L, 128, cin, c.XRES, nullptr, 128, 0,
       conv_b, nullptr, 1.f, nullptr, nullptr, 0);

  // LN1 -> bf16 A for in_proj
  k_ln<<<cdiv(L, 8), 256, 0, c.st>>>(c.XRES, L, 128, norm_g, norm_b,
                                     nullptr, c.ABUF, nullptr, 0, 0);

  // in_proj: xz[L,512]
  cvtT(c, in_proj_w, c.WINP, 128, 512);
  gemm(c, c.ABUF, 128, c.WINP, 128, L, 512, 128, c.XZ, nullptr, 512, 0,
       nullptr, nullptr, 1.f, nullptr, nullptr, 0);

  // depthwise conv + SiLU on xi; build 4-direction xs
  k_dwconv_silu<<<cdiv((long)L * 256, 256), 256, 0, c.st>>>(c.XZ, dw_w, dw_b, c.XC, H, Wd);
  k_prep_xs<<<cdiv(4L * L * 256, 256), 256, 0, c.st>>>(c.XC, c.XS, H, Wd);

  // x_proj padded to N=32 (cols 0..7 dt-rank, 8..15 B, 16..23 C, 24..31 zero)
  k_prep_wxp<<<cdiv(4L * 32 * 256, 256), 256, 0, c.st>>>(x_proj_w, c.WXP);
  for (int k = 0; k < 4; k++){
    gemm(c, c.XS + (long)k * L * 256, 256, c.WXP + (long)k * 32 * 256, 256,
         L, 32, 256, c.XDBL + (long)k * L * 32, nullptr, 32, 0,
         nullptr, nullptr, 1.f, nullptr, nullptr, 0);
  }
  // dt_proj: pad K=8 -> 32 on both A and W so the GEMM stays on the fast path
  k_pad_cols_bf16<<<cdiv(4L * 256 * 32, 256), 256, 0, c.st>>>(dt_proj_w, 8, 8, c.WDTP, 32, 4L * 256);
  for (int k = 0; k < 4; k++){
    k_pad_cols_bf16<<<cdiv((long)L * 32, 256), 256, 0, c.st>>>(c.XDBL + (long)k * L * 32, 32, 8,
                                                               c.DTA, 32, L);
    gemm(c, c.DTA, 32, c.WDTP + (long)k * 256 * 32, 32,
         L, 256, 32, c.DTS + (long)k * L * 256, nullptr, 256, 0,
         nullptr, nullptr, 1.f, nullptr, nullptr, 0);
  }

  // selective scan + direction combine
  k_scan<<<4, 256, 0, c.st>>>(c.DTS, c.XS, c.XDBL, A_log, Ds, dt_bias, c.Y, L);
  k_combine<<<cdiv((long)L * 256, 256), 256, 0, c.st>>>(c.Y, c.YM, H, Wd);

  // out-norm LN * silu(z) -> bf16 [L,256]
  k_ln<<<cdiv(L, 8), 256, 0, c.st>>>(c.YM, L, 256, out_norm_g, out_norm_b,
                                     nullptr, c.YG, c.XZ, 512, 256);

  // out_proj + residual(xres) -> x2
  cvtT(c, out_proj_w, c.WOUTP, 256, 128);
  gemm(c, c.YG, 256, c.WOUTP, 256, L, 128, 256, c.X2, nullptr, 128, 0,
       nullptr, nullptr, 1.f, nullptr, c.XRES, 0);

  // MLP: LN2 -> gelu(x@w1+b1) -> @w2+b2 + residual(x2)
  k_ln<<<cdiv(L, 8), 256, 0, c.st>>>(c.X2, L, 128, norm2_g, norm2_b,
                                     nullptr, c.ABUF, nullptr, 0, 0);
  cvtT(c, mlp_w1, c.WM1, 128, 512);
  gemm(c, c.ABUF, 128, c.WM1, 128, L, 512, 128, nullptr, c.HM, 512, 0,
       mlp_b1, nullptr, 1.f, nullptr, nullptr, 2);
  cvtT(c, mlp_w2, c.WM2, 512, 128);
  gemm(c, c.HM, 512, c.WM2, 512, L, 128, 512, outQ, nullptr, 128, 0,
       mlp_b2, nullptr, 1.f, nullptr, c.X2, 0);
}

static const float BN_SMUL_H = 0.99999500003749971f; // 1/sqrt(1+1e-5)

static void run_stage(Ctx& c, const float* pre, const float* post,
                      int H, int Wd, int cin, int pbA, int pbB, int pbC, int pbF,
                      float* outP)
{
  const long HW = (long)H * Wd;
  // q1: concat channels
  k_prep_A<<<cdiv(HW * 2 * cin, 256), 256, 0, c.st>>>(pre, post, c.ABUF, H, Wd, cin, 0);
  run_st_block(c, pbA, H, Wd, 2 * cin, c.Q1);
  // q2: interleave W
  k_prep_A<<<cdiv(HW * 2 * cin, 256), 256, 0, c.st>>>(pre, post, c.ABUF, H, Wd, cin, 1);
  run_st_block(c, pbB, H, 2 * Wd, cin, c.Q2);
  // q3: concat W
  k_prep_A<<<cdiv(HW * 2 * cin, 256), 256, 0, c.st>>>(pre, post, c.ABUF, H, Wd, cin, 2);
  run_st_block(c, pbC, H, 2 * Wd, cin, c.Q3);
  // fuse: 640->128 1x1 conv + BN + relu, NCHW store
  k_prep_fuse<<<cdiv(HW * 640, 256), 256, 0, c.st>>>(c.Q1, c.Q2, c.Q3, c.ABUF, H, Wd);
  const float* fw  = (const float*)c.din[pbF + 0];
  const float* fb  = (const float*)c.din[pbF + 1];
  const float* fg  = (const float*)c.din[pbF + 2];
  const float* fbe = (const float*)c.din[pbF + 3];
  cvt(c, fw, c.WF, 128L * 640);
  gemm(c, c.ABUF, 640, c.WF, 640, (int)HW, 128, 640, outP, nullptr, (int)HW, 1,
       fb, fg, BN_SMUL_H, fbe, nullptr, 1);
}

static void run_resblock(Ctx& c, const float* xin, int H, int Wd, int pb, float* out)
{
  const long HW = (long)H * Wd;
  const float* w1 = (const float*)c.din[pb + 0];
  const float* g1 = (const float*)c.din[pb + 1];
  const float* b1 = (const float*)c.din[pb + 2];
  const float* w2 = (const float*)c.din[pb + 3];
  const float* g2 = (const float*)c.din[pb + 4];
  const float* b2 = (const float*)c.din[pb + 5];
  k_im2col<<<cdiv(HW * 1152, 256), 256, 0, c.st>>>(xin, c.ABUF, H, Wd);
  cvt(c, w1, c.WSM, 128L * 1152);
  gemm(c, c.ABUF, 1152, c.WSM, 1152, (int)HW, 128, 1152, c.T2, nullptr, (int)HW, 1,
       nullptr, g1, BN_SMUL_H, b1, nullptr, 1);
  k_im2col<<<cdiv(HW * 1152, 256), 256, 0, c.st>>>(c.T2, c.ABUF, H, Wd);
  cvt(c, w2, c.WSM, 128L * 1152);
  gemm(c, c.ABUF, 1152, c.WSM, 1152, (int)HW, 128, 1152, out, nullptr, (int)HW, 1,
       nullptr, g2, BN_SMUL_H, b2, xin, 1);
}

extern "C" void kernel_launch(void* const* d_in, const int* in_sizes, int n_in,
                              void* d_out, int out_size, void* d_ws, size_t ws_size,
                              hipStream_t stream)
{
  (void)in_sizes; (void)n_in; (void)out_size; (void)ws_size;
  Ctx c; c.st = stream; c.din = d_in;

  // workspace layout (reused sequentially; sized for Lmax = 56*112 = 6272)
  char* base = (char*)d_ws;
  size_t off = 0;
  auto take = [&](size_t bytes)->char*{
    char* p = base + off;
    off += (bytes + 255) & ~(size_t)255;
    return p;
  };
  const long LMAX = 6272;
  c.XRES = (float*)take(LMAX * 128 * 4);
  c.ABUF = (bf16_t*)take(3136L * 1152 * 2);       // max of prep/fuse/im2col A
  c.XZ   = (float*)take(LMAX * 512 * 4);
  c.XC   = (float*)take(LMAX * 256 * 4);
  c.XS   = (bf16_t*)take(4 * LMAX * 256 * 2);
  c.XDBL = (float*)take(4 * LMAX * 32 * 4);
  c.DTS  = (float*)take(4 * LMAX * 256 * 4);
  c.Y    = (float*)take(4 * 256 * LMAX * 4);
  c.YM   = (float*)take(LMAX * 256 * 4);
  c.YG   = (bf16_t*)take(LMAX * 256 * 2);
  c.X2   = (float*)take(LMAX * 128 * 4);
  c.HM   = (bf16_t*)take(LMAX * 512 * 2);
  c.DTA  = (bf16_t*)take(LMAX * 32 * 2);
  c.Q1   = (float*)take(LMAX * 128 * 4);
  c.Q2   = (float*)take(LMAX * 128 * 4);
  c.Q3   = (float*)take(LMAX * 128 * 4);
  c.P4   = (float*)take(128L * 3136 * 4);
  c.P3   = (float*)take(128L * 3136 * 4);
  c.P2   = (float*)take(128L * 3136 * 4);
  c.PF   = (float*)take(128L * 3136 * 4);
  c.T1   = (float*)take(128L * 3136 * 4);
  c.T2   = (float*)take(128L * 3136 * 4);
  c.WCONV= (bf16_t*)take(128L * 1024 * 2);
  c.WINP = (bf16_t*)take(512L * 128 * 2);
  c.WXP  = (bf16_t*)take(4L * 32 * 256 * 2);
  c.WDTP = (bf16_t*)take(4L * 256 * 32 * 2);
  c.WOUTP= (bf16_t*)take(128L * 256 * 2);
  c.WM1  = (bf16_t*)take(512L * 128 * 2);
  c.WM2  = (bf16_t*)take(128L * 512 * 2);
  c.WF   = (bf16_t*)take(128L * 640 * 2);
  c.WSM  = (bf16_t*)take(128L * 1152 * 2);

  const float* pre1  = (const float*)d_in[0];
  const float* pre2  = (const float*)d_in[1];
  const float* pre3  = (const float*)d_in[2];
  const float* pre4  = (const float*)d_in[3];
  const float* post1 = (const float*)d_in[4];
  const float* post2 = (const float*)d_in[5];
  const float* post3 = (const float*)d_in[6];
  const float* post4 = (const float*)d_in[7];

  // parameter flattening (dict insertion order):
  //   st['41','42','43','31','32','33','21','22','23','11','12','13'] x 21 arrays
  //   fuse['4','3','2','1'] x 4 arrays ; smooth['3','2','1'] x 6 arrays
  auto stb = [](int i){ return 8 + i * 21; };
  const int FU4 = 260, FU3 = 264, FU2 = 268, FU1 = 272;
  const int SM3 = 276, SM2 = 282, SM1 = 288;

  // stage 4 (7x7, cin 512) and stage 3 (14x14, cin 256)
  run_stage(c, pre4, post4, 7, 7, 512, stb(0), stb(1), stb(2), FU4, c.P4);
  run_stage(c, pre3, post3, 14, 14, 256, stb(3), stb(4), stb(5), FU3, c.P3);
  k_upsample_add<<<cdiv(128L * 14 * 14, 256), 256, 0, stream>>>(c.P4, 7, 7, c.P3, c.T1, 14, 14);
  run_resblock(c, c.T1, 14, 14, SM3, c.P3);

  // stage 2 (28x28, cin 128)
  run_stage(c, pre2, post2, 28, 28, 128, stb(6), stb(7), stb(8), FU2, c.P2);
  k_upsample_add<<<cdiv(128L * 28 * 28, 256), 256, 0, stream>>>(c.P3, 14, 14, c.P2, c.T1, 28, 28);
  run_resblock(c, c.T1, 28, 28, SM2, c.P2);

  // stage 1 (56x56, cin 64) -> final output
  run_stage(c, pre1, post1, 56, 56, 64, stb(9), stb(10), stb(11), FU1, c.PF);
  k_upsample_add<<<cdiv(128L * 56 * 56, 256), 256, 0, stream>>>(c.P2, 28, 28, c.PF, c.T1, 56, 56);
  run_resblock(c, c.T1, 56, 56, SM1, (float*)d_out);
}
